// Time_Mix_44341242364302
// MI455X (gfx1250) — matrix operand order repaired and checked by reading, not compiled, not run on hardware
//
#include <hip/hip_runtime.h>
#include <hip/hip_bf16.h>
#include <math.h>

// ---------------------------------------------------------------------------
// RWKV time-mix, MI455X (gfx1250, wave32, WMMA 16x16x32 f16)
// B=16, H=32, D=64, E=2048, TS_H=32, N_BRANCH=5
//
// All WMMA A/B operands are kept in "fragment-ready" layout: for a 16x32 f16
// K-step, lane L (half=L>>4, mr=L&15) owns 16 contiguous halves, so fragment
// loads are two b128 loads instead of 16 scalar u16 gathers.
//   A element (m=b, k_local): half= k_local>>4,    e=  k_local&15
//   B element (k_local, n):   half= k_local>>4,    e=  k_local&15
// ---------------------------------------------------------------------------

#define BB   16
#define HH   32
#define DD   64
#define EE   2048
#define TSH  32
#define NBR  5
#define NLAY 24

typedef __attribute__((ext_vector_type(16))) _Float16 v16h;
typedef __attribute__((ext_vector_type(8)))  float    v8f;

union h16u { v16h v; _Float16 h[16]; };
union f8u  { v8f  v; float    f[8];  };

// fragment-slot index for an A-matrix element (b = row 0..15, k = column)
__device__ __forceinline__ int a_slot(int b, int k) {
    int kk = k >> 5, kl = k & 31;
    int half = kl >> 4;
    int e    = kl & 15;
    int lane = half * 16 + b;
    return (kk * 32 + lane) * 16 + e;                  // halves
}

// ---------------------------------------------------------------------------
// Kernel 1: diff, gate-input (f16, fragment-ready), new_state slice
// ---------------------------------------------------------------------------
__global__ void prep_k(const float* __restrict__ x, const float* __restrict__ state,
                       const float* __restrict__ gmu, const int* __restrict__ lidp,
                       float* __restrict__ diff, _Float16* __restrict__ xm16,
                       float* __restrict__ out_state) {
    int i = blockIdx.x * 256 + threadIdx.x;            // < 16*2048
    int b = i >> 11, e = i & 2047;
    int lid = *lidp;
    float xv = x[i];
    float lx = state[((size_t)lid * BB + b) * EE + e];
    float d  = xv - lx;
    diff[i]  = d;
    float gm = gmu[e];
    xm16[a_slot(b, e)] = (_Float16)(xv * gm + lx * (1.0f - gm));
    out_state[((size_t)lid * BB + b) * EE + e] = xv;
}

// ---------------------------------------------------------------------------
// Kernel 2: hdn = tanh((x + diff*mu) @ Aw + Ab)   [5,16,32]; 1 wave per (s,b)
// ---------------------------------------------------------------------------
__global__ void hdn_k(const float* __restrict__ x, const float* __restrict__ diff,
                      const float* __restrict__ mus, const float* __restrict__ Aw,
                      const float* __restrict__ Ab, float* __restrict__ hdn) {
    int s = blockIdx.x >> 4, b = blockIdx.x & 15;
    int h = threadIdx.x;                               // 0..31
    float acc = Ab[s * TSH + h];
    const float* xr = x + b * EE;
    const float* dr = diff + b * EE;
    const float* mr = mus + s * EE;
    const float* aw = Aw + (size_t)s * EE * TSH + h;
    for (int e = 0; e < EE; ++e) {
        float li = xr[e] + dr[e] * mr[e];
        acc += li * aw[(size_t)e * TSH];
    }
    hdn[(s * BB + b) * TSH + h] = tanhf(acc);
}

// ---------------------------------------------------------------------------
// Kernel 3: lora -> ts (f16, fragment-ready per (s,h) tile)  [5,16,2048]
// ts_sw layout: (((s*32+h)*2 + kk)*32 + lane)*16 + e
// ---------------------------------------------------------------------------
__global__ void lora_ts_k(const float* __restrict__ x, const float* __restrict__ diff,
                          const float* __restrict__ hdn, const float* __restrict__ Bw,
                          const float* __restrict__ Bb, const float* __restrict__ lambdas,
                          _Float16* __restrict__ ts_sw) {
    int t = blockIdx.x * 256 + threadIdx.x;            // < 5*16*2048
    int s = t >> 15, b = (t >> 11) & 15, e = t & 2047;
    float acc = Bb[s * EE + e] + lambdas[s * EE + e];
    const float* hd = hdn + (s * BB + b) * TSH;
    const float* bw = Bw + (size_t)s * TSH * EE + e;
    #pragma unroll 8
    for (int h = 0; h < TSH; ++h) acc += hd[h] * bw[(size_t)h * EE];
    float ts = acc * diff[b * EE + e] + x[b * EE + e];
    int hh = e >> 6, d = e & 63;
    ts_sw[(size_t)(s * HH + hh) * 1024 + a_slot(b, d)] = (_Float16)ts;
}

// ---------------------------------------------------------------------------
// Kernel 4: per-head projection via WMMA. One wave per (branch s, head h):
//   proj[s,:,h,:] = ts[s,:,h,:] (16x64) @ MW[s,h] (64x64) + Mb[s,h]
// ---------------------------------------------------------------------------
__global__ void proj_k(const _Float16* __restrict__ ts_sw, const float* __restrict__ MW,
                       const float* __restrict__ Mb, float* __restrict__ proj) {
    int s = blockIdx.x >> 5, h = blockIdx.x & 31;
    __shared__ _Float16 Bs[2 * 4 * 32 * 16];           // fragment-ready, 8 KB
    int lane = threadIdx.x;                            // 0..31

    const float* mw = MW + (size_t)(s * HH + h) * DD * DD;
    for (int i = lane; i < DD * DD; i += 32) {         // coalesced read, swizzled write
        int d = i >> 6, n = i & 63;
        int nt = n >> 4, mr = n & 15;
        int kk = d >> 5, kl = d & 31;
        int half = kl >> 4, e = kl & 15;
        Bs[(((kk * 4 + nt) * 32) + half * 16 + mr) * 16 + e] = (_Float16)mw[i];
    }
    __syncthreads();

    const _Float16* ab = ts_sw + (size_t)(s * HH + h) * 1024;
    int half = lane >> 4, mr = lane & 15;
    f8u acc[4];
    #pragma unroll
    for (int nt = 0; nt < 4; ++nt)
        for (int r = 0; r < 8; ++r) acc[nt].f[r] = 0.0f;

    #pragma unroll
    for (int kk = 0; kk < 2; ++kk) {
        h16u au;
        au.v = *(const v16h*)(ab + (kk * 32 + lane) * 16);        // 2x b128
        #pragma unroll
        for (int nt = 0; nt < 4; ++nt) {
            h16u bu;
            bu.v = *(const v16h*)(Bs + ((kk * 4 + nt) * 32 + lane) * 16);  // 2x ds_b128
            acc[nt].v = __builtin_amdgcn_wmma_f32_16x16x32_f16(
                false, au.v, false, bu.v, (short)0, acc[nt].v, false, false);
        }
    }
    #pragma unroll
    for (int nt = 0; nt < 4; ++nt)
        #pragma unroll
        for (int r = 0; r < 8; ++r) {
            int m = r + half * 8;
            int e = nt * 16 + mr;
            proj[(((size_t)(s * BB + m)) * HH + h) * DD + e] =
                acc[nt].f[r] + Mb[(s * HH + h) * DD + e];
        }
}

// ---------------------------------------------------------------------------
// Kernel 5/7: out[16,2048] = A16 @ W[2048,2048] + bias  (WMMA)
// A16 is fragment-ready in global (L2-resident, 64 KB). Per K-step the block
// cooperatively stages the 32x128 W tile (coalesced) fragment-ready in LDS.
// 8 waves/block, 1 N-tile per wave, grid 16 -> 128 N-tiles.
// ---------------------------------------------------------------------------
__global__ void gemm16_k(const _Float16* __restrict__ A16, const float* __restrict__ W,
                         const float* __restrict__ bias, float* __restrict__ out) {
    __shared__ _Float16 Bs[8 * 32 * 16];               // fragment-ready, 8 KB
    int tid  = threadIdx.x;
    int wave = tid >> 5, lane = tid & 31;
    int half = lane >> 4, mr = lane & 15;
    int nblk = blockIdx.x * 128;
    int n0   = nblk + wave * 16;

    f8u acc;
    #pragma unroll
    for (int r = 0; r < 8; ++r) acc.f[r] = 0.0f;

    for (int kk = 0; kk < 64; ++kk) {
        int k0 = kk * 32;
        // stage W[k0:k0+32, nblk:nblk+128] -> Bs (coalesced rows of 512 B)
        #pragma unroll
        for (int j = 0; j < 16; ++j) {
            int idx = tid + j * 256;                   // < 4096
            int n = idx & 127, kl = idx >> 7;
            float wv = W[(size_t)(k0 + kl) * EE + nblk + n];
            if (kk < 63)                               // prefetch next K tile
                __builtin_prefetch(&W[(size_t)(k0 + 32 + kl) * EE + nblk + n], 0, 3);
            int w_ = n >> 4, m_ = n & 15;
            int h_ = kl >> 4, e_ = kl & 15;
            Bs[((w_ * 32) + h_ * 16 + m_) * 16 + e_] = (_Float16)wv;
        }
        __syncthreads();

        h16u au, bu;
        au.v = *(const v16h*)(A16 + (kk * 32 + lane) * 16);       // 2x b128 (L2)
        bu.v = *(const v16h*)(Bs + (wave * 32 + lane) * 16);      // 2x ds_b128
        acc.v = __builtin_amdgcn_wmma_f32_16x16x32_f16(
            false, au.v, false, bu.v, (short)0, acc.v, false, false);
        __syncthreads();
    }
    #pragma unroll
    for (int r = 0; r < 8; ++r) {
        int m = r + half * 8;
        out[(size_t)m * EE + n0 + mr] = acc.f[r] + bias[n0 + mr];
    }
}

// ---------------------------------------------------------------------------
// Kernel 6: WKV recurrence + new_S + group-norm + silu gate. 1 block per (b,h).
// rwkv_i = u_i*k_i*(v.r) + (S r)_i ; new_S = w_i*S + k_i*v_j
// ng16 written fragment-ready for the output GEMM.
// ---------------------------------------------------------------------------
__global__ void wkv_k(const float* __restrict__ proj, const float* __restrict__ smat_in,
                      const int* __restrict__ lidp, const float* __restrict__ g,
                      const float* __restrict__ gamma, const float* __restrict__ beta,
                      float* __restrict__ out_smat, _Float16* __restrict__ ng16) {
    int b = blockIdx.x >> 5, h = blockIdx.x & 31;
    int i = threadIdx.x;                               // 0..63
    __shared__ float rr[DD], rv[DD], red[DD];

    const int SBE = BB * HH * DD;                      // per-branch stride
    int base = (b * HH + h) * DD + i;
    float kk_ = proj[0 * SBE + base];
    float r_  = proj[1 * SBE + base];
    float wr  = proj[2 * SBE + base];
    float v_  = proj[3 * SBE + base];
    float u_  = proj[4 * SBE + base];
    float wi  = expf(-expf(wr));
    rr[i] = r_; rv[i] = v_;
    __syncthreads();

    // v . r
    red[i] = rv[i] * rr[i]; __syncthreads();
    for (int st = 32; st > 0; st >>= 1) { if (i < st) red[i] += red[i + st]; __syncthreads(); }
    float vr = red[0]; __syncthreads();

    int lid = *lidp;
    size_t sbase = (((size_t)lid * BB + b) * HH + h) * (DD * DD);
    const float4* srow = (const float4*)(smat_in + sbase + i * DD);
    float4*       drow = (float4*)(out_smat + sbase + i * DD);
    float acc = 0.0f;
    #pragma unroll 4
    for (int j4 = 0; j4 < DD / 4; ++j4) {
        float4 s4 = srow[j4];
        int j = j4 * 4;
        acc += s4.x * rr[j] + s4.y * rr[j + 1] + s4.z * rr[j + 2] + s4.w * rr[j + 3];
        float4 n4;
        n4.x = wi * s4.x + kk_ * rv[j];
        n4.y = wi * s4.y + kk_ * rv[j + 1];
        n4.z = wi * s4.z + kk_ * rv[j + 2];
        n4.w = wi * s4.w + kk_ * rv[j + 3];
        drow[j4] = n4;
    }
    float rw = u_ * kk_ * vr + acc;

    // group norm over the 64 channels of this head
    red[i] = rw; __syncthreads();
    for (int st = 32; st > 0; st >>= 1) { if (i < st) red[i] += red[i + st]; __syncthreads(); }
    float mean = red[0] * (1.0f / DD); __syncthreads();
    red[i] = rw * rw; __syncthreads();
    for (int st = 32; st > 0; st >>= 1) { if (i < st) red[i] += red[i + st]; __syncthreads(); }
    float var = red[0] * (1.0f / DD) - mean * mean;

    int ch = h * DD + i;
    float nm = (rw - mean) * rsqrtf(var + 1e-3f) * gamma[ch] + beta[ch];
    float gv = g[b * EE + ch];
    float sg = gv / (1.0f + expf(-gv));               // silu
    ng16[a_slot(b, ch)] = (_Float16)(nm * sg);
}

// ---------------------------------------------------------------------------
// Launch
// ---------------------------------------------------------------------------
extern "C" void kernel_launch(void* const* d_in, const int* in_sizes, int n_in,
                              void* d_out, int out_size, void* d_ws, size_t ws_size,
                              hipStream_t stream) {
    const float* x       = (const float*)d_in[0];
    const float* state   = (const float*)d_in[1];
    const float* smat    = (const float*)d_in[2];
    const float* mus     = (const float*)d_in[3];
    const float* lambdas = (const float*)d_in[4];
    const float* Aw      = (const float*)d_in[5];
    const float* Ab      = (const float*)d_in[6];
    const float* Bw      = (const float*)d_in[7];
    const float* Bb      = (const float*)d_in[8];
    const float* MW      = (const float*)d_in[9];
    const float* Mb      = (const float*)d_in[10];
    const float* gmu     = (const float*)d_in[11];
    const float* gW      = (const float*)d_in[12];
    const float* gb      = (const float*)d_in[13];
    const float* gn_g    = (const float*)d_in[14];
    const float* gn_b    = (const float*)d_in[15];
    const float* oW      = (const float*)d_in[16];
    const float* ob      = (const float*)d_in[17];
    const int*   lid     = (const int*)d_in[18];

    float* out       = (float*)d_out;
    float* out_state = out + (size_t)BB * EE;                       // after time_mixed
    float* out_smat  = out_state + (size_t)NLAY * BB * EE;

    char* w = (char*)d_ws;
    float*    diff  = (float*)(w + 0);                              // 131072 B
    _Float16* xm16  = (_Float16*)(w + 131072);                      //  65536 B
    _Float16* ts_sw = (_Float16*)(w + 196608);                      // 327680 B
    float*    hdn   = (float*)(w + 524288);                         //  10240 B
    float*    proj  = (float*)(w + 534528);                         // 655360 B
    float*    gbuf  = (float*)(w + 1189888);                        // 131072 B
    _Float16* ng16  = (_Float16*)(w + 1320960);                     //  65536 B

    // Bulk-copy untouched layers of the state outputs; live slices overwritten below.
    hipMemcpyAsync(out_state, state, (size_t)NLAY * BB * EE * sizeof(float),
                   hipMemcpyDeviceToDevice, stream);
    hipMemcpyAsync(out_smat, smat, (size_t)NLAY * BB * HH * DD * DD * sizeof(float),
                   hipMemcpyDeviceToDevice, stream);

    prep_k<<<(BB * EE) / 256, 256, 0, stream>>>(x, state, gmu, lid, diff, xm16, out_state);
    hdn_k<<<NBR * BB, 32, 0, stream>>>(x, diff, mus, Aw, Ab, hdn);
    lora_ts_k<<<(NBR * BB * EE) / 256, 256, 0, stream>>>(x, diff, hdn, Bw, Bb, lambdas, ts_sw);
    proj_k<<<NBR * HH, 32, 0, stream>>>(ts_sw, MW, Mb, proj);
    gemm16_k<<<16, 256, 0, stream>>>(xm16, gW, gb, gbuf);           // gate projection
    wkv_k<<<BB * HH, 64, 0, stream>>>(proj, smat, lid, gbuf, gn_g, gn_b, out_smat, ng16);
    gemm16_k<<<16, 256, 0, stream>>>(ng16, oW, ob, out);            // output projection
}
